// LSTM_52905407152294
// MI455X (gfx1250) — matrix operand order repaired and checked by reading, not compiled, not run on hardware
//
#include <hip/hip_runtime.h>
#include <math.h>

#define SEQ   512
#define DIN   256
#define HID   1024
#define CLS   1024
#define BATCH 128

// LDS staging layout: per-gate regions with 16B pad per 1024B of stream
// (matches TDM pad_interval=1024B, pad_amount=16B).
#define WX_G  8320u          // 16*256*2 + 8*16
#define WH_G  33280u         // 16*1024*2 + 32*16
#define SMEM_BYTES (4u * WX_G + 4u * WH_G)   // 166400 B < 320KB WGP LDS

typedef __attribute__((ext_vector_type(16))) __bf16 v16bf;
typedef __attribute__((ext_vector_type(8)))  float  v8f;
typedef __attribute__((ext_vector_type(4)))  unsigned int v4u;
typedef __attribute__((ext_vector_type(8)))  int v8i;
typedef __attribute__((ext_vector_type(4)))  int v4i;

#if defined(__has_builtin)
#if __has_builtin(__builtin_amdgcn_tensor_load_to_lds)
#define USE_TDM 1
#endif
#endif

union FragU { uint4 u[2]; v16bf v; };

// A-fragment (16x32 bf16, row-major): lane = row (lane&15);
// K contiguous [k+hi*16,+16) (matches B fragments)
__device__ __forceinline__ v16bf ld_a(const __bf16* __restrict__ row, int k, int hi) {
    FragU f;
    f.u[0] = *reinterpret_cast<const uint4*>(row + k + hi * 16);
    f.u[1] = *reinterpret_cast<const uint4*>(row + k + hi * 16 + 8);
    return f.v;
}

// B-fragment from LDS: 32 contiguous bytes (16 bf16 K-values), 16B aligned.
__device__ __forceinline__ v16bf ld_b_lds(const char* p) {
    const uint4* q = reinterpret_cast<const uint4*>(p);
    FragU f;
    f.u[0] = q[0];
    f.u[1] = q[1];
    return f.v;
}

__device__ __forceinline__ float sigmoidf_(float x) { return 1.0f / (1.0f + __expf(-x)); }

#ifdef USE_TDM
// TDM: 2-D tile (rows x kLen bf16) from row-major global into padded LDS region.
__device__ __forceinline__ void tdm_load_2d(unsigned ldsOff, const __bf16* gptr,
                                            unsigned kLen, unsigned rows) {
    unsigned long long ga = (unsigned long long)(uintptr_t)gptr;
    v4u g0;
    g0.x = 1u;                                              // count=1, user mode
    g0.y = ldsOff;                                          // lds_addr
    g0.z = (unsigned)ga;                                    // global_addr[31:0]
    g0.w = (unsigned)((ga >> 32) & 0x1FFFFFFu) | 0x80000000u; // ga[56:32] | type=2
    v8i g1;
    g1[0] = (int)0x07D10000u;   // data_size=2B, pad_enable, interval=1024B, amount=4 DW
    g1[1] = (int)(kLen << 16);  // tensor_dim0[15:0]  @bits63:48
    g1[2] = (int)(rows << 16);  // tensor_dim1[15:0]  @bits95:80
    g1[3] = (int)(kLen << 16);  // tile_dim0          @bits127:112
    g1[4] = (int)rows;          // tile_dim1          @bits143:128
    g1[5] = (int)kLen;          // tensor_dim0_stride @bits207:160 (low 32)
    g1[6] = 0;
    g1[7] = 0;
    v4i gz4 = {0, 0, 0, 0};
    v8i gz8 = {0, 0, 0, 0, 0, 0, 0, 0};
    __builtin_amdgcn_tensor_load_to_lds(g0, g1, gz4, gz4, gz8, 0);
}
#else
// Fallback: cooperative copy with identical padded layout.
__device__ __forceinline__ void stage_region_fallback(char* smem, unsigned base,
                                                      const __bf16* __restrict__ g,
                                                      int K, int tid) {
    const int chunksPerRow = K / 8;
    const int total = 16 * chunksPerRow;
    for (int cI = tid; cI < total; cI += 256) {
        int row = cI / chunksPerRow;
        int kk  = (cI - row * chunksPerRow) * 8;
        unsigned s   = (unsigned)(row * K + kk) * 2u;
        unsigned off = s + ((s >> 10) << 4);
        *reinterpret_cast<uint4*>(smem + base + off) =
            *reinterpret_cast<const uint4*>(g + (size_t)row * K + kk);
    }
}
#endif

// ---------------- prep kernels ----------------

__global__ void cvt_bf16(const float* __restrict__ in, __bf16* __restrict__ out, int n) {
    int i = blockIdx.x * blockDim.x + threadIdx.x;
    if (i < n) out[i] = (__bf16)in[i];
}

__global__ void pack4T(const float* __restrict__ w0, const float* __restrict__ w1,
                       const float* __restrict__ w2, const float* __restrict__ w3,
                       __bf16* __restrict__ out, int K) {
    int idx = blockIdx.x * blockDim.x + threadIdx.x;
    int total = 4 * HID * K;
    if (idx >= total) return;
    int e = idx / K, k = idx - e * K;
    int gate = e >> 10, colw = e & (HID - 1);
    const float* w = gate == 0 ? w0 : gate == 1 ? w1 : gate == 2 ? w2 : w3;
    out[idx] = (__bf16)w[(size_t)k * HID + colw];
}

__global__ void pack1T(const float* __restrict__ w, __bf16* __restrict__ out, int K, int N) {
    int idx = blockIdx.x * blockDim.x + threadIdx.x;
    if (idx >= K * N) return;
    int n = idx / K, k = idx - n * K;
    out[idx] = (__bf16)w[(size_t)k * N + n];
}

__global__ void init_state(__bf16* __restrict__ h0, float* __restrict__ c,
                           const float* __restrict__ Bg, const float* __restrict__ Bi,
                           const float* __restrict__ Bf2, const float* __restrict__ Bo,
                           float* __restrict__ bx) {
    int i = blockIdx.x * blockDim.x + threadIdx.x;
    if (i < BATCH * HID) { h0[i] = (__bf16)0.0f; c[i] = 0.0f; }
    if (i < HID) {
        bx[0 * HID + i] = Bg[i];
        bx[1 * HID + i] = Bi[i];
        bx[2 * HID + i] = Bf2[i];
        bx[3 * HID + i] = Bo[i];
    }
}

// ---------------- recurrent step ----------------
// grid: 64 blocks (16 hidden cols each); block: 8 wave32s, wave w = batch rows [16w,16w+16).
// Weight slice (4 gates x 16 cols x 1280 K bf16) is TDM-staged into LDS once per step.
__global__ __launch_bounds__(256)
void lstm_step(const __bf16* __restrict__ xbf,   // [B][T][D]
               const __bf16* __restrict__ WxT,   // [4H][D]
               const __bf16* __restrict__ WhT,   // [4H][H]
               const float*  __restrict__ bx,    // [4H]
               const __bf16* __restrict__ hprev, // [B][H]
               __bf16*       __restrict__ hnext, // [B][H]
               float*        __restrict__ c,     // [B][H]
               int t)
{
    extern __shared__ char smem[];
    const int lane = threadIdx.x & 31;
    const int wave = threadIdx.x >> 5;
    const int rl   = lane & 15;
    const int hi   = lane >> 4;
    const int m0   = wave * 16;
    const int n0   = blockIdx.x * 16;

#ifdef USE_TDM
    if (wave == 0) {
        unsigned lbase = (unsigned)(size_t)smem;   // low 32 bits = LDS offset
#pragma unroll
        for (int g = 0; g < 4; ++g) {
            tdm_load_2d(lbase + g * WX_G, WxT + (size_t)(g * HID + n0) * DIN, DIN, 16);
            tdm_load_2d(lbase + 4 * WX_G + g * WH_G,
                        WhT + (size_t)(g * HID + n0) * HID, HID, 16);
        }
        __builtin_amdgcn_s_wait_tensorcnt(0);
    }
#else
    {
        const int tid = threadIdx.x;
#pragma unroll
        for (int g = 0; g < 4; ++g) {
            stage_region_fallback(smem, g * WX_G, WxT + (size_t)(g * HID + n0) * DIN, DIN, tid);
            stage_region_fallback(smem, 4 * WX_G + g * WH_G,
                                  WhT + (size_t)(g * HID + n0) * HID, HID, tid);
        }
    }
#endif
    __syncthreads();

    // Per-lane LDS fragment bases (column = rl). Padded stream: off = s + (s>>10)*16.
    const unsigned wxb = (unsigned)rl * 512u + (unsigned)(rl >> 1) * 16u;
    const unsigned whb = 4 * WX_G + (unsigned)rl * 2080u;

    v8f acc0 = {}, acc1 = {}, acc2 = {}, acc3 = {};

    { // x projection, K = DIN
        const __bf16* arow = xbf + ((size_t)(m0 + rl) * SEQ + t) * DIN;
        v16bf a = ld_a(arow, 0, hi);
        for (int k = 0; k < DIN; k += 32) {
            int kn = (k + 32 < DIN) ? (k + 32) : 0;
            v16bf an = ld_a(arow, kn, hi);                 // prefetch next A
            const unsigned ko2 = (unsigned)(k + hi * 16) * 2u;  // < 512: no pad crossing
            // batch all 4 B-fragment loads before the WMMA burst
            v16bf b0 = ld_b_lds(smem + 0u * WX_G + wxb + ko2);
            v16bf b1 = ld_b_lds(smem + 1u * WX_G + wxb + ko2);
            v16bf b2 = ld_b_lds(smem + 2u * WX_G + wxb + ko2);
            v16bf b3 = ld_b_lds(smem + 3u * WX_G + wxb + ko2);
            acc0 = __builtin_amdgcn_wmma_f32_16x16x32_bf16(false, a, false, b0, (short)0, acc0, false, false);
            acc1 = __builtin_amdgcn_wmma_f32_16x16x32_bf16(false, a, false, b1, (short)0, acc1, false, false);
            acc2 = __builtin_amdgcn_wmma_f32_16x16x32_bf16(false, a, false, b2, (short)0, acc2, false, false);
            acc3 = __builtin_amdgcn_wmma_f32_16x16x32_bf16(false, a, false, b3, (short)0, acc3, false, false);
            a = an;
        }
    }
    { // recurrence, K = HID
        const __bf16* arow = hprev + (size_t)(m0 + rl) * HID;
        v16bf a = ld_a(arow, 0, hi);
        for (int k = 0; k < HID; k += 32) {
            int kn = (k + 32 < HID) ? (k + 32) : 0;
            v16bf an = ld_a(arow, kn, hi);                 // prefetch next A
            const unsigned ko  = (unsigned)(k + hi * 16);
            const unsigned off = ko * 2u + ((ko >> 9) << 4);
            v16bf b0 = ld_b_lds(smem + whb + 0u * WH_G + off);
            v16bf b1 = ld_b_lds(smem + whb + 1u * WH_G + off);
            v16bf b2 = ld_b_lds(smem + whb + 2u * WH_G + off);
            v16bf b3 = ld_b_lds(smem + whb + 3u * WH_G + off);
            acc0 = __builtin_amdgcn_wmma_f32_16x16x32_bf16(false, a, false, b0, (short)0, acc0, false, false);
            acc1 = __builtin_amdgcn_wmma_f32_16x16x32_bf16(false, a, false, b1, (short)0, acc1, false, false);
            acc2 = __builtin_amdgcn_wmma_f32_16x16x32_bf16(false, a, false, b2, (short)0, acc2, false, false);
            acc3 = __builtin_amdgcn_wmma_f32_16x16x32_bf16(false, a, false, b3, (short)0, acc3, false, false);
            a = an;
        }
    }

    // fused gates + cell update. C/D layout: VGPR r -> row = m0 + hi*8 + r, col = n0 + rl.
    const int col = n0 + rl;
    const float bg = bx[0 * HID + col];
    const float bi = bx[1 * HID + col];
    const float bf = bx[2 * HID + col];
    const float bo = bx[3 * HID + col];
#pragma unroll
    for (int r = 0; r < 8; ++r) {
        const int row = m0 + hi * 8 + r;
        const size_t off = (size_t)row * HID + col;
        float gv = tanhf(acc0[r] + bg);
        float iv = sigmoidf_(acc1[r] + bi);
        float fv = sigmoidf_(acc2[r] + bf);
        float ov = sigmoidf_(acc3[r] + bo);
        float cn = gv * iv + c[off] * fv;
        c[off] = cn;
        hnext[off] = (__bf16)(tanhf(cn) * ov);
    }
}

// ---------------- final projection ----------------
__global__ __launch_bounds__(256)
void lstm_proj(const __bf16* __restrict__ h,   // [B][H]
               const __bf16* __restrict__ WpT, // [CLS][H]
               const float*  __restrict__ Bp,  // [CLS]
               float* __restrict__ out)        // [B][CLS]
{
    const int lane = threadIdx.x & 31;
    const int wave = threadIdx.x >> 5;
    const int rl = lane & 15, hi = lane >> 4;
    const int m0 = wave * 16;
    const int n0 = blockIdx.x * 16;

    v8f acc = {};
    const __bf16* arow = h + (size_t)(m0 + rl) * HID;
    const __bf16* bcol = WpT + (size_t)(n0 + rl) * HID;
    for (int k = 0; k < HID; k += 32) {
        FragU fb;
        fb.u[0] = *reinterpret_cast<const uint4*>(bcol + k + hi * 16);
        fb.u[1] = *reinterpret_cast<const uint4*>(bcol + k + hi * 16 + 8);
        v16bf a = ld_a(arow, k, hi);
        acc = __builtin_amdgcn_wmma_f32_16x16x32_bf16(
            false, a, false, fb.v, (short)0, acc, false, false);
    }
    const int col = n0 + rl;
    const float bias = Bp[col];
#pragma unroll
    for (int r = 0; r < 8; ++r) {
        const int row = m0 + hi * 8 + r;
        out[(size_t)row * CLS + col] = acc[r] + bias;
    }
}

extern "C" void kernel_launch(void* const* d_in, const int* in_sizes, int n_in,
                              void* d_out, int out_size, void* d_ws, size_t ws_size,
                              hipStream_t stream) {
    (void)in_sizes; (void)n_in; (void)out_size; (void)ws_size;

    const float* x   = (const float*)d_in[0];
    const float* Wgx = (const float*)d_in[1];
    const float* Wix = (const float*)d_in[2];
    const float* Wfx = (const float*)d_in[3];
    const float* Wox = (const float*)d_in[4];
    const float* Wgh = (const float*)d_in[5];
    const float* Wih = (const float*)d_in[6];
    const float* Wfh = (const float*)d_in[7];
    const float* Woh = (const float*)d_in[8];
    const float* Bg  = (const float*)d_in[9];
    const float* Bi  = (const float*)d_in[10];
    const float* Bf  = (const float*)d_in[11];
    const float* Bo  = (const float*)d_in[12];
    const float* Wph = (const float*)d_in[13];
    const float* Bp  = (const float*)d_in[14];

    char* p = (char*)d_ws;
    auto alloc = [&](size_t bytes) {
        char* r = p;
        p += (bytes + 255) & ~(size_t)255;
        return r;
    };
    __bf16* xbf = (__bf16*)alloc((size_t)BATCH * SEQ * DIN * 2); // 33.5 MB
    __bf16* WxT = (__bf16*)alloc((size_t)4 * HID * DIN * 2);     // 2 MB
    __bf16* WhT = (__bf16*)alloc((size_t)4 * HID * HID * 2);     // 8 MB (L2-resident)
    __bf16* WpT = (__bf16*)alloc((size_t)CLS * HID * 2);         // 2 MB
    float*  bx  = (float*) alloc((size_t)4 * HID * 4);
    __bf16* h0  = (__bf16*)alloc((size_t)BATCH * HID * 2);
    __bf16* h1  = (__bf16*)alloc((size_t)BATCH * HID * 2);
    float*  c   = (float*) alloc((size_t)BATCH * HID * 4);

    const int nx = BATCH * SEQ * DIN;
    cvt_bf16<<<(nx + 255) / 256, 256, 0, stream>>>(x, xbf, nx);
    const int nwx = 4 * HID * DIN;
    pack4T<<<(nwx + 255) / 256, 256, 0, stream>>>(Wgx, Wix, Wfx, Wox, WxT, DIN);
    const int nwh = 4 * HID * HID;
    pack4T<<<(nwh + 255) / 256, 256, 0, stream>>>(Wgh, Wih, Wfh, Woh, WhT, HID);
    const int nwp = CLS * HID;
    pack1T<<<(nwp + 255) / 256, 256, 0, stream>>>(Wph, WpT, HID, CLS);
    init_state<<<(BATCH * HID + 255) / 256, 256, 0, stream>>>(h0, c, Bg, Bi, Bf, Bo, bx);

    for (int t = 0; t < SEQ; ++t) {
        __bf16* hp = (t & 1) ? h1 : h0;
        __bf16* hn = (t & 1) ? h0 : h1;
        lstm_step<<<HID / 16, 256, SMEM_BYTES, stream>>>(xbf, WxT, WhT, bx, hp, hn, c, t);
    }
    // t=511 reads h1, writes h0 -> final hidden state in h0
    lstm_proj<<<CLS / 16, 256, 0, stream>>>(h0, WpT, Bp, (float*)d_out);
}
